// PathfinderBlock_72859825209819
// MI455X (gfx1250) — 16-bit and 8-bit matrix operand order repaired and checked by reading, compiled, not run on hardware
//
#include <hip/hip_runtime.h>
#include <math.h>

constexpr int B   = 32;
constexpr int C   = 512;
constexpr int L   = 1024;
constexpr int LP  = L + 2;
constexpr int TOK = B * L;

typedef __attribute__((ext_vector_type(16))) __bf16 v16bf;
typedef __attribute__((ext_vector_type(8)))  float  v8f;
typedef __attribute__((ext_vector_type(8)))  int    v8i;

union AFrag16 { v16bf v; uint4 q[2]; };
union AFrag8  { v8i   v; uint2 q[4]; };
union BFrag8  { v8i   v; uint4 q[2]; };

__device__ __forceinline__ unsigned short f2bf(float f) {
  union { float f; unsigned u; } x; x.f = f;
  unsigned r = x.u + 0x7FFFu + ((x.u >> 16) & 1u);
  return (unsigned short)(r >> 16);
}
__device__ __forceinline__ float bf2f(unsigned short h) {
  union { unsigned u; float f; } x; x.u = ((unsigned)h) << 16;
  return x.f;
}
__device__ __forceinline__ float gelu_exact(float x) {
  return 0.5f * x * (1.0f + erff(x * 0.70710678118654752f));
}

constexpr size_t OFF_XT    = 0;
constexpr size_t OFF_CONV  = OFF_XT   + (size_t)B * LP * C * 2;
constexpr size_t OFF_XQ    = OFF_CONV + (size_t)TOK * C * 2;
constexpr size_t OFF_TSC   = OFF_XQ   + (size_t)TOK * C;
constexpr size_t OFF_WK    = OFF_TSC  + (size_t)TOK * 4;
constexpr size_t OFF_WQ    = OFF_WK   + (size_t)3 * C * C * 2;
constexpr size_t OFF_BNS   = OFF_WQ   + (size_t)C * C;
constexpr size_t OFF_BNB   = OFF_BNS  + (size_t)C * 4;
constexpr size_t OFF_PART  = OFF_BNB  + (size_t)C * 4;
constexpr size_t OFF_MEANW = OFF_PART + 256 * 4;

__global__ void k_prep_wk(const float* __restrict__ conv_w, unsigned short* __restrict__ wk) {
  int i = blockIdx.x * 256 + threadIdx.x;
  #pragma unroll
  for (int k = 0; k < 3; ++k)
    wk[(size_t)k * C * C + i] = f2bf(conv_w[(size_t)i * 3 + k]);
}

__global__ void k_absmean_part(const float* __restrict__ w, float* __restrict__ part) {
  __shared__ float sm[256];
  int tid = threadIdx.x;
  float s = 0.f;
  int base = blockIdx.x * 1024;
  for (int i = base + tid; i < base + 1024; i += 256) s += fabsf(w[i]);
  sm[tid] = s; __syncthreads();
  for (int o = 128; o > 0; o >>= 1) { if (tid < o) sm[tid] += sm[tid + o]; __syncthreads(); }
  if (tid == 0) part[blockIdx.x] = sm[0];
}
__global__ void k_absmean_final(const float* __restrict__ part, float* __restrict__ meanw) {
  __shared__ float sm[256];
  int tid = threadIdx.x;
  sm[tid] = part[tid]; __syncthreads();
  for (int o = 128; o > 0; o >>= 1) { if (tid < o) sm[tid] += sm[tid + o]; __syncthreads(); }
  if (tid == 0) *meanw = fmaxf(sm[0] / (float)(C * C), 1e-5f);
}
__global__ void k_quant_w(const float* __restrict__ w, const float* __restrict__ meanw,
                          signed char* __restrict__ wq) {
  int i = blockIdx.x * 256 + threadIdx.x;
  float s = 1.0f / *meanw;
  float q = rintf(w[i] * s);
  q = fminf(fmaxf(q, -1.f), 1.f);
  wq[i] = (signed char)q;
}

__global__ void k_transpose(const float* __restrict__ x, unsigned short* __restrict__ xT) {
  __shared__ float tile[32][33];
  int blk  = blockIdx.x;
  int lt   = blk % (L / 32);
  int rest = blk / (L / 32);
  int ct   = rest % (C / 32);
  int bb   = rest / (C / 32);
  int tx = threadIdx.x & 31, ty = threadIdx.x >> 5;
  int c0 = ct * 32, l0 = lt * 32;
  #pragma unroll
  for (int j = 0; j < 32; j += 8)
    tile[ty + j][tx] = x[((size_t)bb * C + (c0 + ty + j)) * L + l0 + tx];
  __syncthreads();
  #pragma unroll
  for (int j = 0; j < 32; j += 8)
    xT[((size_t)bb * LP + (l0 + ty + j + 1)) * C + c0 + tx] = f2bf(tile[tx][ty + j]);
}
__global__ void k_zero_pad(unsigned short* __restrict__ xT) {
  int i = blockIdx.x * 256 + threadIdx.x;
  int bb = i / (2 * C); int rc = i % (2 * C);
  size_t row = (rc / C) ? (size_t)(LP - 1) : 0;
  xT[((size_t)bb * LP + row) * C + (rc % C)] = 0;
}

__global__ __launch_bounds__(256, 2)
void k_conv_gemm(const unsigned short* __restrict__ xT, const unsigned short* __restrict__ wk,
                 const float* __restrict__ conv_b, unsigned short* __restrict__ convout) {
  int tid = threadIdx.x;
  int wave = tid >> 5, lane = tid & 31;
  int hf = lane >> 4, lm = lane & 15;
  int nb = blockIdx.x % (C / 64);
  int mb = blockIdx.x / (C / 64);
  int M0 = mb * 128 + (wave & 3) * 32;
  int N0 = nb * 64  + (wave >> 2) * 32;
  int bb = M0 / L;
  int l0 = M0 % L;

  v8f acc[2][2] = {};
  for (int k = 0; k < 3; ++k) {
    const unsigned short* xbase = xT + ((size_t)bb * LP + l0 + k) * C;
    const unsigned short* wbase = wk + (size_t)k * C * C;
    for (int kk = 0; kk < C; kk += 32) {
      AFrag16 a[2], bf[2];
      #pragma unroll
      for (int mt = 0; mt < 2; ++mt) {
        const unsigned short* p = xbase + (size_t)(mt * 16 + lm) * C + kk;
        a[mt].q[0] = *(const uint4*)(p + hf * 8);
        a[mt].q[1] = *(const uint4*)(p + 16 + hf * 8);
      }
      #pragma unroll
      for (int nt = 0; nt < 2; ++nt) {
        const unsigned short* p = wbase + (size_t)(N0 + nt * 16 + lm) * C + kk + hf * 8;
        bf[nt].q[0] = *(const uint4*)(p);
        bf[nt].q[1] = *(const uint4*)(p + 16);
      }
      #pragma unroll
      for (int mt = 0; mt < 2; ++mt)
        #pragma unroll
        for (int nt = 0; nt < 2; ++nt)
          acc[mt][nt] = __builtin_amdgcn_wmma_f32_16x16x32_bf16(
              false, a[mt].v, false, bf[nt].v, (short)0, acc[mt][nt], false, false);
    }
  }
  #pragma unroll
  for (int nt = 0; nt < 2; ++nt) {
    int o = N0 + nt * 16 + lm;
    float bias = conv_b[o];
    #pragma unroll
    for (int mt = 0; mt < 2; ++mt)
      #pragma unroll
      for (int r = 0; r < 8; ++r) {
        int tok = M0 + mt * 16 + hf * 8 + r;
        convout[(size_t)tok * C + o] = f2bf(acc[mt][nt][r] + bias);
      }
  }
}

__global__ void k_bn_stats(const unsigned short* __restrict__ convout,
                           const float* __restrict__ gamma, const float* __restrict__ beta,
                           float* __restrict__ bns, float* __restrict__ bnb) {
  __shared__ float ss[256], sq[256];
  int c = blockIdx.x, tid = threadIdx.x;
  float s = 0.f, q = 0.f;
  for (int t = tid; t < TOK; t += 256) {
    float v = bf2f(convout[(size_t)t * C + c]);
    s += v; q += v * v;
  }
  ss[tid] = s; sq[tid] = q; __syncthreads();
  for (int o = 128; o > 0; o >>= 1) {
    if (tid < o) { ss[tid] += ss[tid + o]; sq[tid] += sq[tid + o]; }
    __syncthreads();
  }
  if (tid == 0) {
    float mean = ss[0] / (float)TOK;
    float var  = sq[0] / (float)TOK - mean * mean;
    float sc   = gamma[c] * rsqrtf(var + 1e-5f);
    bns[c] = sc;
    bnb[c] = beta[c] - mean * sc;
  }
}

__global__ void k_quant_act(const unsigned short* __restrict__ convout,
                            const float* __restrict__ bns, const float* __restrict__ bnb,
                            signed char* __restrict__ xq, float* __restrict__ tokscale) {
  int wave = threadIdx.x >> 5, lane = threadIdx.x & 31;
  int t = blockIdx.x * 8 + wave;
  const unsigned short* row = convout + (size_t)t * C;
  float g[16];
  float amax = 0.f;
  #pragma unroll
  for (int ch = 0; ch < 2; ++ch) {
    int c0 = ch * 256 + lane * 8;
    uint4 qv = *(const uint4*)(row + c0);
    unsigned u[4] = { qv.x, qv.y, qv.z, qv.w };
    #pragma unroll
    for (int d = 0; d < 4; ++d)
      #pragma unroll
      for (int h = 0; h < 2; ++h) {
        int c = c0 + d * 2 + h;
        float v = bf2f((unsigned short)((u[d] >> (16 * h)) & 0xFFFFu));
        float gg = gelu_exact(v * bns[c] + bnb[c]);
        g[ch * 8 + d * 2 + h] = gg;
        amax = fmaxf(amax, fabsf(gg));
      }
  }
  for (int o = 16; o > 0; o >>= 1) amax = fmaxf(amax, __shfl_xor(amax, o));
  amax = fmaxf(amax, 1e-5f);
  float s = 127.f / amax;
  #pragma unroll
  for (int ch = 0; ch < 2; ++ch) {
    unsigned p0 = 0, p1 = 0;
    #pragma unroll
    for (int j = 0; j < 8; ++j) {
      float q = fminf(fmaxf(rintf(g[ch * 8 + j] * s), -128.f), 127.f);
      unsigned byte = (unsigned)((int)q & 0xFF);
      if (j < 4) p0 |= byte << (8 * j); else p1 |= byte << (8 * (j - 4));
    }
    *(uint2*)(xq + (size_t)t * C + ch * 256 + lane * 8) = make_uint2(p0, p1);
  }
  if (lane == 0) tokscale[t] = amax / 127.f;
}

__global__ __launch_bounds__(256, 2)
void k_bitlinear_gemm(const signed char* __restrict__ xq, const signed char* __restrict__ wq,
                      const float* __restrict__ tokscale, const float* __restrict__ meanw,
                      const float* __restrict__ x, float* __restrict__ out) {
  int tid = threadIdx.x;
  int wave = tid >> 5, lane = tid & 31;
  int hf = lane >> 4, lm = lane & 15;
  int nb = blockIdx.x % (C / 64);
  int mb = blockIdx.x / (C / 64);
  int M0 = mb * 128 + (wave & 3) * 32;
  int N0 = nb * 64  + (wave >> 2) * 32;
  int bb = M0 / L;
  int l0 = M0 % L;

  v8i acc[2][2] = {};
  for (int kk = 0; kk < C; kk += 64) {
    AFrag8 a[2]; BFrag8 bf[2];
    #pragma unroll
    for (int mt = 0; mt < 2; ++mt) {
      const signed char* p = xq + (size_t)(M0 + mt * 16 + lm) * C + kk;
      int b0 = hf * 16;
      a[mt].q[0] = *(const uint2*)(p + b0);
      a[mt].q[1] = *(const uint2*)(p + b0 + 8);
      a[mt].q[2] = *(const uint2*)(p + b0 + 32);
      a[mt].q[3] = *(const uint2*)(p + b0 + 40);
    }
    #pragma unroll
    for (int nt = 0; nt < 2; ++nt) {
      const signed char* p = wq + (size_t)(N0 + nt * 16 + lm) * C + kk + hf * 16;
      bf[nt].q[0] = *(const uint4*)(p);
      bf[nt].q[1] = *(const uint4*)(p + 32);
    }
    #pragma unroll
    for (int mt = 0; mt < 2; ++mt)
      #pragma unroll
      for (int nt = 0; nt < 2; ++nt)
        acc[mt][nt] = __builtin_amdgcn_wmma_i32_16x16x64_iu8(
            true, a[mt].v, true, bf[nt].v, acc[mt][nt], false, false);
  }

  float wsc = *meanw;
  #pragma unroll
  for (int nt = 0; nt < 2; ++nt) {
    int o = N0 + nt * 16 + lm;
    #pragma unroll
    for (int mt = 0; mt < 2; ++mt)
      #pragma unroll
      for (int r = 0; r < 8; ++r) {
        int m  = mt * 16 + hf * 8 + r;
        int tok = M0 + m;
        float deq = tokscale[tok] * wsc;
        float f = (float)acc[mt][nt][r] * deq;
        float g = gelu_exact(f);
        size_t idx = ((size_t)bb * C + o) * L + (size_t)(l0 + m);
        out[idx] = g + x[idx];
      }
  }
}

extern "C" void kernel_launch(void* const* d_in, const int* in_sizes, int n_in,
                              void* d_out, int out_size, void* d_ws, size_t ws_size,
                              hipStream_t stream) {
  const float* x       = (const float*)d_in[0];
  const float* conv_w  = (const float*)d_in[1];
  const float* conv_b  = (const float*)d_in[2];
  const float* gamma   = (const float*)d_in[3];
  const float* beta    = (const float*)d_in[4];
  const float* proj_w  = (const float*)d_in[5];
  float* out = (float*)d_out;

  char* ws = (char*)d_ws;
  unsigned short* xT    = (unsigned short*)(ws + OFF_XT);
  unsigned short* conv  = (unsigned short*)(ws + OFF_CONV);
  signed char*    xq    = (signed char*)  (ws + OFF_XQ);
  float*          tsc   = (float*)        (ws + OFF_TSC);
  unsigned short* wk    = (unsigned short*)(ws + OFF_WK);
  signed char*    wq    = (signed char*)  (ws + OFF_WQ);
  float*          bns   = (float*)        (ws + OFF_BNS);
  float*          bnb   = (float*)        (ws + OFF_BNB);
  float*          part  = (float*)        (ws + OFF_PART);
  float*          meanw = (float*)        (ws + OFF_MEANW);

  k_prep_wk     <<<(C * C) / 256, 256, 0, stream>>>(conv_w, wk);
  k_absmean_part<<<256, 256, 0, stream>>>(proj_w, part);
  k_absmean_final<<<1, 256, 0, stream>>>(part, meanw);
  k_quant_w     <<<(C * C) / 256, 256, 0, stream>>>(proj_w, meanw, wq);

  k_zero_pad    <<<(B * 2 * C) / 256, 256, 0, stream>>>(xT);
  k_transpose   <<<B * (C / 32) * (L / 32), 256, 0, stream>>>(x, xT);

  k_conv_gemm   <<<(TOK / 128) * (C / 64), 256, 0, stream>>>(xT, wk, conv_b, conv);

  k_bn_stats    <<<C, 256, 0, stream>>>(conv, gamma, beta, bns, bnb);
  k_quant_act   <<<TOK / 8, 256, 0, stream>>>(conv, bns, bnb, xq, tsc);

  k_bitlinear_gemm<<<(TOK / 128) * (C / 64), 256, 0, stream>>>(xq, wq, tsc, meanw, x, out);
}
